// GCN_47339129536790
// MI455X (gfx1250) — matrix operand order repaired and checked by reading, not compiled, not run on hardware
//
#include <hip/hip_runtime.h>
#include <hip/hip_bf16.h>

// ---- problem constants (match reference) ----
#define NN 50000
#define EE 800000
#define FF 64
#define HH 128

typedef __attribute__((ext_vector_type(16))) __bf16 v16bf;
typedef __attribute__((ext_vector_type(8)))  __bf16 v8bf;
typedef __attribute__((ext_vector_type(8)))  float  v8f;

// ---------------------------------------------------------------------------
// small utility kernels
// ---------------------------------------------------------------------------
__global__ void cvt_bf16_kernel(const float* __restrict__ s, __bf16* __restrict__ d, int n) {
    int i = blockIdx.x * blockDim.x + threadIdx.x;
    if (i < n) d[i] = (__bf16)s[i];
}

__global__ void deg_init_kernel(float* __restrict__ deg, int n) {
    int i = blockIdx.x * blockDim.x + threadIdx.x;
    if (i < n) deg[i] = 1.0f;            // self-loop
}

__global__ void deg_accum_kernel(const int* __restrict__ dst, float* __restrict__ deg, int e) {
    int i = blockIdx.x * blockDim.x + threadIdx.x;
    if (i < e) unsafeAtomicAdd(&deg[dst[i]], 1.0f);
}

__global__ void rsqrt_kernel(float* __restrict__ deg, int n) {
    int i = blockIdx.x * blockDim.x + threadIdx.x;
    if (i < n) deg[i] = rsqrtf(deg[i]);
}

__global__ void zero_kernel(float4* __restrict__ p, int n4) {
    int i = blockIdx.x * blockDim.x + threadIdx.x;
    if (i < n4) p[i] = make_float4(0.f, 0.f, 0.f, 0.f);
}

// agg[dst] += hw[src] * dinv[src]*dinv[dst], edges + implicit self-loops.
// 32 lanes per edge, 4 floats per lane (H = 128).
__global__ void scatter_kernel(const float* __restrict__ hw,
                               const int* __restrict__ src, const int* __restrict__ dst,
                               const float* __restrict__ dinv,
                               float* __restrict__ agg, int etot) {
    int t = blockIdx.x * blockDim.x + threadIdx.x;
    int e = t >> 5;
    if (e >= etot) return;
    int fo = (t & 31) * 4;
    int s, d;
    if (e < EE) { s = src[e]; d = dst[e]; }
    else        { s = e - EE; d = s; }
    float norm = dinv[s] * dinv[d];
    const float4 v = *(const float4*)(hw + (size_t)s * HH + fo);
    float* o = agg + (size_t)d * HH + fo;
    unsafeAtomicAdd(o + 0, v.x * norm);
    unsafeAtomicAdd(o + 1, v.y * norm);
    unsafeAtomicAdd(o + 2, v.z * norm);
    unsafeAtomicAdd(o + 3, v.w * norm);
}

__global__ void bias_relu_kernel(float* __restrict__ h, const float* __restrict__ b, int total) {
    int i = blockIdx.x * blockDim.x + threadIdx.x;
    if (i < total) h[i] = fmaxf(h[i] + b[i & (HH - 1)], 0.0f);
}

// ---------------------------------------------------------------------------
// WMMA GEMM:  Out[N, HO] = act(A[N, K] @ W[K, HO] + bias)
//   - A: f32 in memory -> bf16 fragments (two float4 loads per 8 elems)
//   - W: bf16, staged TRANSPOSED in LDS with +8 pad per column so each
//     lane's B fragment is two 16B-aligned ds_load_b128's, conflict-free
//   - one wave per 16-column tile (blockDim.x = 32 * HO/16), 16 rows/block
//   - ACT: 0 = none, 1 = relu, 2 = sigmoid (fast v_rcp_f32)
// Requires: rows % 16 == 0, K % 32 == 0, HO % 16 == 0, HO <= 128, K <= 128.
// ---------------------------------------------------------------------------
template <int K, int HO, int ACT, bool BIAS>
__global__ void gemm_wmma_kernel(const float* __restrict__ A,
                                 const __bf16* __restrict__ W,
                                 const float* __restrict__ bias,
                                 float* __restrict__ Out) {
    constexpr int KP = K + 8;                 // padded column stride (bf16)
    __shared__ __bf16 wldsT[HO * KP];

    const int tid = threadIdx.x;
    // stage W transposed: wldsT[c*KP + k] = W[k*HO + c]
    for (int i = tid; i < K * HO; i += 32 * (HO / 16)) {
        int k = i / HO;
        int c = i - k * HO;
        wldsT[c * KP + k] = W[i];
    }
    __syncthreads();

    const int wave = tid >> 5;
    const int lane = tid & 31;
    const int c0   = wave * 16;               // column tile
    const int row0 = blockIdx.x * 16;         // row tile
    const int m    = lane & 15;
    const int half = lane >> 4;
    const int n    = c0 + (lane & 15);

    v8f acc = {};
#pragma unroll
    for (int k0 = 0; k0 < K; k0 += 32) {
        // A fragment (16x32 bf16, ISA layout): lane = row m;
        // elems 0-7 -> K = half*8 + j, elems 8-15 -> K = 16 + half*8 + (j-8)
        const float* arow = A + (size_t)(row0 + m) * K + k0 + half * 8;
        float4 a0 = *(const float4*)(arow + 0);
        float4 a1 = *(const float4*)(arow + 4);
        float4 a2 = *(const float4*)(arow + 16);
        float4 a3 = *(const float4*)(arow + 20);
        v16bf a;
        a[0]  = (__bf16)a0.x; a[1]  = (__bf16)a0.y; a[2]  = (__bf16)a0.z; a[3]  = (__bf16)a0.w;
        a[4]  = (__bf16)a1.x; a[5]  = (__bf16)a1.y; a[6]  = (__bf16)a1.z; a[7]  = (__bf16)a1.w;
        a[8]  = (__bf16)a2.x; a[9]  = (__bf16)a2.y; a[10] = (__bf16)a2.z; a[11] = (__bf16)a2.w;
        a[12] = (__bf16)a3.x; a[13] = (__bf16)a3.y; a[14] = (__bf16)a3.z; a[15] = (__bf16)a3.w;

        // B fragment (32x16 bf16): lane = col n; elems 0-7 -> K = k0 + half*8 + j, elems 8-15 -> K = k0 + 16 + half*8 + (j-8)
        const __bf16* wcol = wldsT + (size_t)n * KP + k0 + half * 8;
        v8bf blo = *(const v8bf*)(wcol);
        v8bf bhi = *(const v8bf*)(wcol + 16);
        v16bf b;
#pragma unroll
        for (int j = 0; j < 8; ++j) { b[j] = blo[j]; b[j + 8] = bhi[j]; }

        acc = __builtin_amdgcn_wmma_f32_16x16x32_bf16(
            /*neg_a=*/false, a, /*neg_b=*/false, b,
            /*c_mod=*/(short)0, acc, /*reuse_a=*/false, /*reuse_b=*/false);
    }

    // C/D layout: VGPR r -> M = r + 8*half, lane -> N.
    const float bv = BIAS ? bias[n] : 0.0f;
#pragma unroll
    for (int r = 0; r < 8; ++r) {
        int   mm = r + 8 * half;
        float v  = acc[r] + bv;
        if (ACT == 1)      v = fmaxf(v, 0.0f);
        else if (ACT == 2) v = __builtin_amdgcn_rcpf(1.0f + __expf(-v));
        Out[(size_t)(row0 + mm) * HO + n] = v;
    }
}

// ---------------------------------------------------------------------------
// host launcher
// ---------------------------------------------------------------------------
extern "C" void kernel_launch(void* const* d_in, const int* in_sizes, int n_in,
                              void* d_out, int out_size, void* d_ws, size_t ws_size,
                              hipStream_t stream) {
    (void)in_sizes; (void)n_in; (void)out_size; (void)ws_size;

    const float* x      = (const float*)d_in[0];
    const int*   eidx   = (const int*)d_in[1];
    const int*   esrc   = eidx;
    const int*   edst   = eidx + EE;
    const float* enc_w1 = (const float*)d_in[2];
    const float* enc_b1 = (const float*)d_in[3];
    const float* enc_w2 = (const float*)d_in[4];
    const float* enc_b2 = (const float*)d_in[5];
    const float* w_c1   = (const float*)d_in[6];
    const float* b_c1   = (const float*)d_in[7];
    const float* w_c2   = (const float*)d_in[8];
    const float* b_c2   = (const float*)d_in[9];
    const float* w_c3   = (const float*)d_in[10];
    const float* b_c3   = (const float*)d_in[11];
    const float* dec_w1 = (const float*)d_in[12];
    const float* dec_b1 = (const float*)d_in[13];
    const float* dec_w2 = (const float*)d_in[14];
    const float* dec_b2 = (const float*)d_in[15];

    const size_t nh = (size_t)NN * HH;
    float*  bufA = (float*)d_ws;           // h / agg
    float*  bufB = bufA + nh;              // hw / intermediate
    float*  dinv = bufB + nh;              // degree -> rsqrt(degree)
    __bf16* wb   = (__bf16*)(dinv + NN);
    __bf16* w_e1b = wb;                    // 64*128
    __bf16* w_e2b = w_e1b + FF * HH;       // 128*128
    __bf16* w_c1b = w_e2b + HH * HH;
    __bf16* w_c2b = w_c1b + HH * HH;
    __bf16* w_c3b = w_c2b + HH * HH;
    __bf16* w_d1b = w_c3b + HH * HH;
    __bf16* w_d2b = w_d1b + HH * HH;       // 128*64

    // weight conversion to bf16
    cvt_bf16_kernel<<<(FF*HH + 255)/256, 256, 0, stream>>>(enc_w1, w_e1b, FF*HH);
    cvt_bf16_kernel<<<(HH*HH + 255)/256, 256, 0, stream>>>(enc_w2, w_e2b, HH*HH);
    cvt_bf16_kernel<<<(HH*HH + 255)/256, 256, 0, stream>>>(w_c1,   w_c1b, HH*HH);
    cvt_bf16_kernel<<<(HH*HH + 255)/256, 256, 0, stream>>>(w_c2,   w_c2b, HH*HH);
    cvt_bf16_kernel<<<(HH*HH + 255)/256, 256, 0, stream>>>(w_c3,   w_c3b, HH*HH);
    cvt_bf16_kernel<<<(HH*HH + 255)/256, 256, 0, stream>>>(dec_w1, w_d1b, HH*HH);
    cvt_bf16_kernel<<<(HH*FF + 255)/256, 256, 0, stream>>>(dec_w2, w_d2b, HH*FF);

    // degree normalization (self-loops included): dinv = rsqrt(1 + in-degree)
    deg_init_kernel <<<(NN + 255)/256, 256, 0, stream>>>(dinv, NN);
    deg_accum_kernel<<<(EE + 255)/256, 256, 0, stream>>>(edst, dinv, EE);
    rsqrt_kernel    <<<(NN + 255)/256, 256, 0, stream>>>(dinv, NN);

    const int rowBlocks = NN / 16;   // 3125, exact

    // encoder MLP: bufB = relu(x @ enc_w1 + b1); bufA = bufB @ enc_w2 + b2
    gemm_wmma_kernel<FF, HH, 1, true><<<rowBlocks, 256, 0, stream>>>(x,    w_e1b, enc_b1, bufB);
    gemm_wmma_kernel<HH, HH, 0, true><<<rowBlocks, 256, 0, stream>>>(bufB, w_e2b, enc_b2, bufA);

    // three GCN conv layers
    const __bf16* wcs[3] = {w_c1b, w_c2b, w_c3b};
    const float*  bcs[3] = {b_c1,  b_c2,  b_c3};
    const int etot = EE + NN;
    const int scatterBlocks = (etot * 32 + 255) / 256;
    for (int l = 0; l < 3; ++l) {
        gemm_wmma_kernel<HH, HH, 0, false><<<rowBlocks, 256, 0, stream>>>(bufA, wcs[l], nullptr, bufB);
        zero_kernel<<<((int)(nh/4) + 255)/256, 256, 0, stream>>>((float4*)bufA, (int)(nh/4));
        scatter_kernel<<<scatterBlocks, 256, 0, stream>>>(bufB, esrc, edst, dinv, bufA, etot);
        bias_relu_kernel<<<((int)nh + 255)/256, 256, 0, stream>>>(bufA, bcs[l], (int)nh);
    }

    // decoder MLP + sigmoid, final layer writes d_out ([N,64] f32)
    gemm_wmma_kernel<HH, HH, 1, true><<<rowBlocks, 256, 0, stream>>>(bufA, w_d1b, dec_b1, bufB);
    gemm_wmma_kernel<HH, FF, 2, true><<<rowBlocks, 128, 0, stream>>>(bufB, w_d2b, dec_b2, (float*)d_out);
}
